// GraphiT_GT_Layer_21388937134337
// MI455X (gfx1250) — matrix operand order repaired and checked by reading, not compiled, not run on hardware
//
#include <hip/hip_runtime.h>
#include <hip/hip_bf16.h>
#include <cmath>

typedef __attribute__((ext_vector_type(16))) _Float16 v16h;
typedef __attribute__((ext_vector_type(8)))  float    v8f;
typedef __attribute__((ext_vector_type(4)))  float    v4f;

#define BDIM  16
#define NDIM  256
#define CDIM  64      // IN_DIM == OUT_DIM
#define HH    8
#define DD    8
#define ROWS  (BDIM * NDIM)   // 4096
#define FFDIM 128

// ---------------------------------------------------------------------------
// One-wave 16x16 output tile GEMM step using v_wmma_f32_16x16x32_f16.
// A: MxK row-major f32 (lda), B: KxN row-major f32 (ldb). K = 32*ksteps.
// VGPR layouts used here (A/B k order contiguous, not the ISA table order):
//   A (16x32 f16): element e of v16h holds K = 16*half + e, M = lane%16
//   B (32x16 f16): element e of v16h holds K = 16*half + e, N = lane%16
//   C/D (16x16 f32): VGPR r holds M = r + 8*half, N = lane%16
// ---------------------------------------------------------------------------
__device__ __forceinline__ v8f wmma_tile(const float* __restrict__ A, int lda, int row0,
                                         const float* __restrict__ B, int ldb, int col0,
                                         int ksteps, v8f c)
{
    const int lane   = threadIdx.x & 31;
    const int mn     = lane & 15;
    const int half   = lane >> 4;
    const int kbaseA = half * 8;
    const int kHalfB = half * 16;
#pragma unroll
    for (int ks = 0; ks < 4; ++ks) {
        if (ks >= ksteps) break;
        const int k0 = ks * 32;
        v16h a, b;
#pragma unroll
        for (int e = 0; e < 16; ++e) {
            int kA = k0 + kHalfB + e;
            a[e] = (_Float16)A[(row0 + mn) * lda + kA];
            int kB = k0 + kHalfB + e;
            b[e] = (_Float16)B[kB * ldb + (col0 + mn)];
        }
        c = __builtin_amdgcn_wmma_f32_16x16x32_f16(false, a, false, b,
                                                   (short)0, c, false, false);
    }
    return c;
}

// ---------------------------------------------------------------------------
// LayerNorm: one 64-thread block per row of 64.
// ---------------------------------------------------------------------------
__global__ __launch_bounds__(64)
void ln_kernel(const float* __restrict__ x, float* __restrict__ y,
               const float* __restrict__ g, const float* __restrict__ b)
{
    __shared__ float s1[64], s2[64];
    const int row = blockIdx.x;
    const int t   = threadIdx.x;
    float v = x[row * CDIM + t];
    s1[t] = v; s2[t] = v * v;
    __syncthreads();
#pragma unroll
    for (int st = 32; st > 0; st >>= 1) {
        if (t < st) { s1[t] += s1[t + st]; s2[t] += s2[t + st]; }
        __syncthreads();
    }
    float mu  = s1[0] * (1.0f / CDIM);
    float var = s2[0] * (1.0f / CDIM) - mu * mu;
    float r   = rsqrtf(var + 1e-5f);
    y[row * CDIM + t] = (v - mu) * r * g[t] + b[t];
}

// ---------------------------------------------------------------------------
// Q/K/V projections: hn(4096x64) @ {Wq,Wk,Wv}(64x64). K scaled by D^-1/2.
// grid = (256 M-tiles, 12), block = 32 (one wave per 16x16 tile).
// ---------------------------------------------------------------------------
__global__ __launch_bounds__(32)
void qkv_kernel(const float* __restrict__ hn,
                const float* __restrict__ Wq, const float* __restrict__ Wk,
                const float* __restrict__ Wv,
                float* __restrict__ Q, float* __restrict__ K, float* __restrict__ V)
{
    const int mt = blockIdx.x, nt = blockIdx.y;
    const int wsel = nt >> 2;
    const int col0 = (nt & 3) * 16;
    const float* W = (wsel == 0) ? Wq : ((wsel == 1) ? Wk : Wv);
    float* out     = (wsel == 0) ? Q  : ((wsel == 1) ? K  : V);
    const float scale = (wsel == 1) ? 0.35355339059327373f : 1.0f; // 8^-0.5

    v8f c = {};
    c = wmma_tile(hn, CDIM, mt * 16, W, CDIM, col0, 2, c);

    const int lane = threadIdx.x & 31;
    const int n = lane & 15, half = lane >> 4;
#pragma unroll
    for (int r = 0; r < 8; ++r) {
        int row = mt * 16 + r + 8 * half;
        out[row * CDIM + col0 + n] = c[r] * scale;
    }
}

// ---------------------------------------------------------------------------
// Streaming attention: one 256-thread block per (b,i) row.
// Fuses qk broadcast + exp(clip(score)) + mask + denom + Σ s*(V+e_val).
// e_att/e_val are 537 MB streamed exactly once (> 192 MB L2) -> non-temporal
// b128 loads so the reusable K/V/Q working set stays L2-resident.
// ---------------------------------------------------------------------------
__global__ __launch_bounds__(256)
void attn_kernel(const float* __restrict__ Q, const float* __restrict__ K,
                 const float* __restrict__ V,
                 const float* __restrict__ e_att, const float* __restrict__ e_val,
                 const float* __restrict__ mask, float* __restrict__ attnOut)
{
    __shared__ float qk[NDIM][HH];
    __shared__ float qrow[CDIM];
    __shared__ float mrow[NDIM];
    __shared__ v4f   denomP[16][16];
    __shared__ v4f   aggP[16][16];

    const int bi = blockIdx.x;          // b*256 + i
    const int b  = bi >> 8;
    const int i  = bi & 255;
    const int t  = threadIdx.x;

    if (t < CDIM) qrow[t] = Q[bi * CDIM + t];
    mrow[t] = mask[b * NDIM + t];
    __syncthreads();

    {   // per-thread: qk[j][h] for j = t, all 8 heads
        const int j = t;
        const float* kr = K + (size_t)(b * NDIM + j) * CDIM;
#pragma unroll
        for (int h = 0; h < HH; ++h) {
            float s = 0.f;
#pragma unroll
            for (int k = 0; k < DD; ++k) s += qrow[h * DD + k] * kr[h * DD + k];
            qk[j][h] = s;
        }
    }
    __syncthreads();

    const int q4   = t & 15;    // which 4-float group of the 64-wide hk row
    const int jsub = t >> 4;    // 16 j's in flight
    const int h    = q4 >> 1;   // 8 consecutive hk (2 groups) share one head
    const float mi = mask[b * NDIM + i];

    const v4f* ea4 = (const v4f*)e_att;
    const v4f* ev4 = (const v4f*)e_val;
    const v4f* v4p = (const v4f*)V;

    v4f den = {0.f, 0.f, 0.f, 0.f};
    v4f agg = {0.f, 0.f, 0.f, 0.f};
    for (int j = jsub; j < NDIM; j += 16) {
        size_t base = ((size_t)bi * NDIM + j) * 16 + q4;   // in v4f units
        v4f ea = __builtin_nontemporal_load(&ea4[base]);   // single-use stream
        v4f ev = __builtin_nontemporal_load(&ev4[base]);   // single-use stream
        v4f vv = v4p[(size_t)(b * NDIM + j) * 16 + q4];    // hot, keep cached
        float qv = qk[j][h];
        float mm = mi * mrow[j];
        float s;
        s = __expf(fminf(fmaxf(qv + ea.x, -5.f), 5.f)) * mm; den.x += s; agg.x += s * (vv.x + ev.x);
        s = __expf(fminf(fmaxf(qv + ea.y, -5.f), 5.f)) * mm; den.y += s; agg.y += s * (vv.y + ev.y);
        s = __expf(fminf(fmaxf(qv + ea.z, -5.f), 5.f)) * mm; den.z += s; agg.z += s * (vv.z + ev.z);
        s = __expf(fminf(fmaxf(qv + ea.w, -5.f), 5.f)) * mm; den.w += s; agg.w += s * (vv.w + ev.w);
    }
    denomP[jsub][q4] = den;
    aggP[jsub][q4]   = agg;
    __syncthreads();

    if (t < 16) {
        v4f d = denomP[0][t], a = aggP[0][t];
#pragma unroll
        for (int s = 1; s < 16; ++s) {
            d += denomP[s][t];
            a += aggP[s][t];
        }
        v4f o;
        o.x = a.x / fmaxf(d.x, 1e-6f);
        o.y = a.y / fmaxf(d.y, 1e-6f);
        o.z = a.z / fmaxf(d.z, 1e-6f);
        o.w = a.w / fmaxf(d.w, 1e-6f);
        ((v4f*)attnOut)[(size_t)bi * 16 + t] = o;
    }
}

// ---------------------------------------------------------------------------
// h1 = h + attnOut @ Wo     (WMMA, residual fused in epilogue)
// ---------------------------------------------------------------------------
__global__ __launch_bounds__(32)
void wo_kernel(const float* __restrict__ attn, const float* __restrict__ Wo,
               const float* __restrict__ h0, float* __restrict__ h1)
{
    const int mt = blockIdx.x, col0 = blockIdx.y * 16;
    v8f c = {};
    c = wmma_tile(attn, CDIM, mt * 16, Wo, CDIM, col0, 2, c);
    const int lane = threadIdx.x & 31;
    const int n = lane & 15, half = lane >> 4;
#pragma unroll
    for (int r = 0; r < 8; ++r) {
        int idx = (mt * 16 + r + 8 * half) * CDIM + col0 + n;
        h1[idx] = h0[idx] + c[r];
    }
}

// ---------------------------------------------------------------------------
// ffa = gelu_exact(hn2 @ Wff1)   (64 -> 128)
// ---------------------------------------------------------------------------
__global__ __launch_bounds__(32)
void ff1_kernel(const float* __restrict__ hn2, const float* __restrict__ Wff1,
                float* __restrict__ ffa)
{
    const int mt = blockIdx.x, col0 = blockIdx.y * 16;
    v8f c = {};
    c = wmma_tile(hn2, CDIM, mt * 16, Wff1, FFDIM, col0, 2, c);
    const int lane = threadIdx.x & 31;
    const int n = lane & 15, half = lane >> 4;
#pragma unroll
    for (int r = 0; r < 8; ++r) {
        int row = mt * 16 + r + 8 * half;
        float x = c[r];
        ffa[row * FFDIM + col0 + n] = 0.5f * x * (1.0f + erff(x * 0.7071067811865476f));
    }
}

// ---------------------------------------------------------------------------
// out = mask * (h1 + ffa @ Wff2)   (128 -> 64, K = 4 WMMA steps)
// ---------------------------------------------------------------------------
__global__ __launch_bounds__(32)
void ff2_kernel(const float* __restrict__ ffa, const float* __restrict__ Wff2,
                const float* __restrict__ h1, const float* __restrict__ mask,
                float* __restrict__ out)
{
    const int mt = blockIdx.x, col0 = blockIdx.y * 16;
    v8f c = {};
    c = wmma_tile(ffa, FFDIM, mt * 16, Wff2, CDIM, col0, 4, c);
    const int lane = threadIdx.x & 31;
    const int n = lane & 15, half = lane >> 4;
#pragma unroll
    for (int r = 0; r < 8; ++r) {
        int row = mt * 16 + r + 8 * half;        // row == b*256 + n -> mask flat index
        int idx = row * CDIM + col0 + n;
        out[idx] = mask[row] * (h1[idx] + c[r]);
    }
}

// ---------------------------------------------------------------------------
extern "C" void kernel_launch(void* const* d_in, const int* in_sizes, int n_in,
                              void* d_out, int out_size, void* d_ws, size_t ws_size,
                              hipStream_t stream)
{
    (void)in_sizes; (void)n_in; (void)out_size; (void)ws_size;

    const float* h     = (const float*)d_in[0];
    // d_in[1] (p) is unused by the reference
    const float* e_att = (const float*)d_in[2];
    const float* e_val = (const float*)d_in[3];
    const float* mask  = (const float*)d_in[4];
    const float* Wq    = (const float*)d_in[5];
    const float* Wk    = (const float*)d_in[6];
    const float* Wv    = (const float*)d_in[7];
    const float* Wo    = (const float*)d_in[8];
    const float* ln1g  = (const float*)d_in[9];
    const float* ln1b  = (const float*)d_in[10];
    const float* ln2g  = (const float*)d_in[11];
    const float* ln2b  = (const float*)d_in[12];
    const float* Wff1  = (const float*)d_in[13];
    const float* Wff2  = (const float*)d_in[14];

    float* ws   = (float*)d_ws;
    float* hn   = ws; ws += (size_t)ROWS * CDIM;
    float* Qb   = ws; ws += (size_t)ROWS * CDIM;
    float* Kb   = ws; ws += (size_t)ROWS * CDIM;
    float* Vb   = ws; ws += (size_t)ROWS * CDIM;
    float* attn = ws; ws += (size_t)ROWS * CDIM;
    float* h1   = ws; ws += (size_t)ROWS * CDIM;
    float* hn2  = ws; ws += (size_t)ROWS * CDIM;
    float* ffa  = ws; ws += (size_t)ROWS * FFDIM;

    ln_kernel  <<<ROWS, 64, 0, stream>>>(h, hn, ln1g, ln1b);
    qkv_kernel <<<dim3(ROWS / 16, 12), 32, 0, stream>>>(hn, Wq, Wk, Wv, Qb, Kb, Vb);
    attn_kernel<<<ROWS, 256, 0, stream>>>(Qb, Kb, Vb, e_att, e_val, mask, attn);
    wo_kernel  <<<dim3(ROWS / 16, 4), 32, 0, stream>>>(attn, Wo, h, h1);
    ln_kernel  <<<ROWS, 64, 0, stream>>>(h1, hn2, ln2g, ln2b);
    ff1_kernel <<<dim3(ROWS / 16, 8), 32, 0, stream>>>(hn2, Wff1, ffa);
    ff2_kernel <<<dim3(ROWS / 16, 4), 32, 0, stream>>>(ffa, Wff2, h1, mask, (float*)d_out);
}
